// SSDBoxCoder_32899449487525
// MI455X (gfx1250) — compile-verified, matrix operand order checked by reading, not run on hardware
//
#include <hip/hip_runtime.h>
#include <hip/hip_bf16.h>
#include <math.h>

#define NPRIOR 49140
#define NB 32
#define NGT 64
#define FG_T 0.6f
#define BG_T 0.4f

#ifndef USE_ASYNC_LDS
#define USE_ASYNC_LDS 1
#endif

typedef __attribute__((ext_vector_type(16))) _Float16 v16h;
typedef __attribute__((ext_vector_type(8)))  float    v8f;

static __device__ __forceinline__ unsigned long long pk_max(unsigned long long x,
                                                            unsigned long long y) {
  return x > y ? x : y;
}

// ---------------------------------------------------------------------------
// Pass 1: per-prior best GT (max over the 64 GTs), cls thresholding, loc encode.
// Wave-tiled: each wave32 owns 16 priors x 64 gts (four 16x16 tiles).
// The union prefactor areaP[i]+areaG[j] for each 16x16 tile is produced with
// v_wmma_f32_16x16x32_f16 as a rank-2 outer-product matmul.
// ---------------------------------------------------------------------------
__global__ __launch_bounds__(256) void k_match(
    const float* __restrict__ gt,   // [B,64,4] xyxy
    const int*   __restrict__ lab,  // [B,64]
    const float* __restrict__ db,   // [N,4] xywh
    float* __restrict__ loc,        // [B,N,4]
    float* __restrict__ cls)        // [B,N]
{
  __shared__ float4   s_gt[NGT];     // gt boxes xyxy
  __shared__ _Float16 s_gah[NGT];    // gt areas * 2^-10 (f16)
  __shared__ int      s_lab[NGT];
  __shared__ float4   s_px[128];     // this block's priors, xyxy
  __shared__ _Float16 s_pah[128];    // prior areas * 2^-10 (f16)

  const int tid = threadIdx.x;
  const int b   = blockIdx.y;

  // ---- stage GT boxes into LDS via CDNA5 async copy (ASYNCcnt-tracked) ----
#if USE_ASYNC_LDS
  {
    unsigned ldsoff = (unsigned)(unsigned long long)((const float*)s_gt + tid);
    const float* src = gt + (size_t)b * NGT * 4 + tid;   // 256 dwords total
    asm volatile("global_load_async_to_lds_b32 %0, %1, off"
                 :: "v"(ldsoff), "v"(src) : "memory");
  }
#else
  if (tid < NGT) s_gt[tid] = ((const float4*)gt)[b * NGT + tid];
#endif
  if (tid < NGT) s_lab[tid] = lab[b * NGT + tid];

  // ---- stage + transform this block's 128 priors (xywh -> xyxy, area) ----
  {
    const int p = blockIdx.x * 128 + tid;
    float4 d = make_float4(0.f, 0.f, 0.f, 0.f);
    if (tid < 128 && p < NPRIOR) d = ((const float4*)db)[p];
    if (tid < 128) {
      s_px[tid]  = make_float4(d.x - 0.5f * d.z, d.y - 0.5f * d.w,
                               d.x + 0.5f * d.z, d.y + 0.5f * d.w);
      s_pah[tid] = (_Float16)(d.z * d.w * (1.0f / 1024.0f));
    }
  }
#if USE_ASYNC_LDS
  asm volatile("s_wait_asynccnt 0" ::: "memory");
#endif
  __syncthreads();
  if (tid < NGT) {
    float4 g = s_gt[tid];
    s_gah[tid] = (_Float16)((g.z - g.x) * (g.w - g.y) * (1.0f / 1024.0f));
  }
  __syncthreads();

  const int wave = tid >> 5;
  const int lane = tid & 31;
  const int grp  = lane >> 4;   // which half of the wave (rows 0-7 vs 8-15)
  const int lcol = lane & 15;   // column of the 16x16 tile this lane owns
  const int prow = wave * 16;   // base row in s_px for this wave

  // Each lane keeps its 8 prior rows (xyxy) in registers (constant-indexed).
  float4 pb[8];
#pragma unroll
  for (int r = 0; r < 8; ++r) pb[r] = s_px[prow + 8 * grp + r];

  // Branchless WMMA operand assembly: always load with an in-bounds index
  // (lcol <= 15), then mask by lane group with selects -- no exec branches,
  // so the four v_wmma ops can issue back-to-back.
  const bool     lo  = (lane < 16);
  const _Float16 z_h = (_Float16)0.0f;
  const _Float16 m1  = lo ? (_Float16)1.0f : z_h;

  // A (16x32 f16): col0 = areaP/1024, col1 = 1.   Lanes 0-15 hold K=0..7.
  const _Float16 pa = s_pah[prow + lcol];
  v16h A = {};
  A[0] = lo ? pa : z_h;
  A[1] = m1;

  _Float16 ga4[4];
#pragma unroll
  for (int t = 0; t < 4; ++t) ga4[t] = s_gah[t * 16 + lcol];

  // S[t] = 16x16 tile of (areaP[i]+areaG[j])/1024 via WMMA outer product.
  v8f S[4];
  const v8f cz = {};
#pragma unroll
  for (int t = 0; t < 4; ++t) {
    v16h Bm = {};
    Bm[0] = m1;
    Bm[1] = lo ? ga4[t] : z_h;
    S[t] = __builtin_amdgcn_wmma_f32_16x16x32_f16(
        false, A, false, Bm, (short)0, cz, false, false);
  }

  // best over gts per prior row: pack (iou_bits<<32)|(63-j) -> max prefers
  // higher iou, then lower j (matches argmax tie-breaking).
  unsigned long long best[8];
#pragma unroll
  for (int r = 0; r < 8; ++r) best[r] = 0ull;

#pragma unroll
  for (int t = 0; t < 4; ++t) {
    const int j = t * 16 + lcol;
    const float4 g = s_gt[j];
#pragma unroll
    for (int r = 0; r < 8; ++r) {
      float ix = fmaxf(fminf(pb[r].z, g.z) - fmaxf(pb[r].x, g.x), 0.f);
      float iy = fmaxf(fminf(pb[r].w, g.w) - fmaxf(pb[r].y, g.y), 0.f);
      float inter = ix * iy;
      float uni = fmaxf(S[t][r] * 1024.0f - inter, 1e-12f);
      float iou = inter > 0.f ? inter / uni : 0.f;
      unsigned long long pk =
          ((unsigned long long)__float_as_uint(iou) << 32) | (unsigned)(63 - j);
      best[r] = pk_max(best[r], pk);
    }
  }

  const size_t ob = (size_t)b * NPRIOR;
#pragma unroll
  for (int r = 0; r < 8; ++r) {
    unsigned long long v = best[r];
    // reduce across the 16-lane group (bit4 preserved => stays in-group)
    v = pk_max(v, __shfl_xor(v, 1, 32));
    v = pk_max(v, __shfl_xor(v, 2, 32));
    v = pk_max(v, __shfl_xor(v, 4, 32));
    v = pk_max(v, __shfl_xor(v, 8, 32));
    if (lcol == r) {                        // one writer lane per row
      const int i = blockIdx.x * 128 + prow + 8 * grp + r;
      if (i < NPRIOR) {
        float bt = __uint_as_float((unsigned)(v >> 32));
        int   j  = 63 - (int)(v & 0xFFFFFFFFu);
        float c;
        if (bt < BG_T)      c = 0.f;
        else if (bt < FG_T) c = -1.f;
        else                c = (float)(s_lab[j] + 1);
        cls[ob + i] = c;
        float4 g = s_gt[j];
        float gcx = 0.5f * (g.x + g.z), gcy = 0.5f * (g.y + g.w);
        float gw = g.z - g.x,           gh = g.w - g.y;
        float pcx = 0.5f * (pb[r].x + pb[r].z), pcy = 0.5f * (pb[r].y + pb[r].w);
        float pw = pb[r].z - pb[r].x,           ph = pb[r].w - pb[r].y;
        ((float4*)loc)[ob + i] =
            make_float4((gcx - pcx) / pw * 10.f, (gcy - pcy) / ph * 10.f,
                        __logf(gw / pw) * 5.f,   __logf(gh / ph) * 5.f);
      }
    }
  }
}

// ---------------------------------------------------------------------------
// Pass 2: per-GT argmax over all priors (one block per (b, j); recompute IoU).
// ---------------------------------------------------------------------------
__global__ __launch_bounds__(256) void k_best_prior(
    const float* __restrict__ gt, const float* __restrict__ db,
    unsigned* __restrict__ win)
{
  const int b = blockIdx.x >> 6;
  const int j = blockIdx.x & 63;
  const float4 g = ((const float4*)gt)[b * NGT + j];
  const float ga = (g.z - g.x) * (g.w - g.y);

  unsigned long long best = 0ull;
  for (int p = threadIdx.x; p < NPRIOR; p += 256) {
    float4 d = ((const float4*)db)[p];
    float px0 = d.x - 0.5f * d.z, py0 = d.y - 0.5f * d.w;
    float px1 = d.x + 0.5f * d.z, py1 = d.y + 0.5f * d.w;
    float ix = fmaxf(fminf(px1, g.z) - fmaxf(px0, g.x), 0.f);
    float iy = fmaxf(fminf(py1, g.w) - fmaxf(py0, g.y), 0.f);
    float inter = ix * iy;
    float iou = inter > 0.f ? inter / (d.z * d.w + ga - inter) : 0.f;
    unsigned long long pk = ((unsigned long long)__float_as_uint(iou) << 32) |
                            (unsigned)(0xFFFFFFFFu - (unsigned)p);
    best = pk_max(best, pk);
  }
  best = pk_max(best, __shfl_xor(best, 16, 32));
  best = pk_max(best, __shfl_xor(best, 8, 32));
  best = pk_max(best, __shfl_xor(best, 4, 32));
  best = pk_max(best, __shfl_xor(best, 2, 32));
  best = pk_max(best, __shfl_xor(best, 1, 32));

  __shared__ unsigned long long red[8];
  if ((threadIdx.x & 31) == 0) red[threadIdx.x >> 5] = best;
  __syncthreads();
  if (threadIdx.x == 0) {
#pragma unroll
    for (int w = 1; w < 8; ++w) best = pk_max(best, red[w]);
    win[b * NGT + j] = 0xFFFFFFFFu - (unsigned)(best & 0xFFFFFFFFu);
  }
}

// ---------------------------------------------------------------------------
// Pass 3: forced matches (best_t := 2.0 at the per-GT winning prior).
// One thread per batch, ascending j => deterministic last-write-wins.
// ---------------------------------------------------------------------------
__global__ void k_force_match(
    const float* __restrict__ gt, const int* __restrict__ lab,
    const float* __restrict__ db, const unsigned* __restrict__ win,
    float* __restrict__ loc, float* __restrict__ cls)
{
  int b = blockIdx.x * blockDim.x + threadIdx.x;
  if (b >= NB) return;
  const size_t ob = (size_t)b * NPRIOR;
  for (int j = 0; j < NGT; ++j) {
    unsigned p = win[b * NGT + j];
    if (p >= NPRIOR) continue;
    float4 g = ((const float4*)gt)[b * NGT + j];
    float4 d = ((const float4*)db)[p];
    float gcx = 0.5f * (g.x + g.z), gcy = 0.5f * (g.y + g.w);
    float gw = g.z - g.x,           gh = g.w - g.y;
    ((float4*)loc)[ob + p] =
        make_float4((gcx - d.x) / d.z * 10.f, (gcy - d.y) / d.w * 10.f,
                    __logf(gw / d.z) * 5.f,   __logf(gh / d.w) * 5.f);
    cls[ob + p] = (float)(lab[b * NGT + j] + 1);  // best_t = 2.0 >= FG_T
  }
}

extern "C" void kernel_launch(void* const* d_in, const int* in_sizes, int n_in,
                              void* d_out, int out_size, void* d_ws, size_t ws_size,
                              hipStream_t stream) {
  (void)in_sizes; (void)n_in; (void)out_size; (void)ws_size;
  const float* gt  = (const float*)d_in[0];   // gt_boxes [32,64,4] f32 (xyxy)
  const int*   lab = (const int*)d_in[1];     // labels   [32,64]   i32
  const float* db  = (const float*)d_in[2];   // default_boxes [49140,4] f32 (xywh)
  float* loc = (float*)d_out;                                   // [32,49140,4]
  float* cls = (float*)d_out + (size_t)NB * NPRIOR * 4;         // [32,49140]
  unsigned* win = (unsigned*)d_ws;                              // [32,64]

  dim3 g1((NPRIOR + 127) / 128, NB);
  k_match<<<g1, 256, 0, stream>>>(gt, lab, db, loc, cls);
  k_best_prior<<<NB * NGT, 256, 0, stream>>>(gt, db, win);
  k_force_match<<<1, NB, 0, stream>>>(gt, lab, db, win, loc, cls);
}
